// EISANImodel_90623809946266
// MI455X (gfx1250) — 8-bit matrix operand order repaired and checked by reading, compiled, not run on hardware
//
#include <hip/hip_runtime.h>
#include <stdint.h>

#define BATCH    512
#define FEATURES 256
#define BITS     8
#define ENC      2048
#define HIDDEN   8192
#define CLASSES  1000
#define THRESH   16

typedef __attribute__((ext_vector_type(8))) int   v8i;
typedef __attribute__((ext_vector_type(2))) float v2f;
typedef __attribute__((ext_vector_type(8))) float v8f;

#if __has_builtin(__builtin_amdgcn_global_load_async_to_lds_b128) && \
    __has_builtin(__builtin_amdgcn_s_wait_asynccnt)
#define USE_ASYNC_LDS 1
typedef int v4i_vs __attribute__((vector_size(16)));
typedef __attribute__((address_space(1))) v4i_vs as1_v4i;
typedef __attribute__((address_space(3))) v4i_vs as3_v4i;
#else
#define USE_ASYNC_LDS 0
#endif

__device__ __forceinline__ int pack4_i8(float a, float b, float c, float d) {
  return ((int)a & 0xff) | (((int)b & 0xff) << 8) |
         (((int)c & 0xff) << 16) | (((int)d & 0xff) << 24);
}

__global__ void encode_kernel(const float* __restrict__ x,
                              uint8_t* __restrict__ a0T) {
  int idx = blockIdx.x * blockDim.x + threadIdx.x;
  if (idx >= BATCH * FEATURES) return;
  int b = idx / FEATURES, f = idx % FEATURES;
  float v = fminf(fmaxf(x[idx], 0.0f), 1.0f);
  int lvl  = (int)rintf(v * 255.0f);
  int gray = lvl ^ (lvl >> 1);
#pragma unroll
  for (int bit = 0; bit < BITS; ++bit)
    a0T[(size_t)(f * BITS + bit) * BATCH + b] = (uint8_t)((gray >> bit) & 1);
}

template <bool WRITE_T>
__global__ __launch_bounds__(128) void binmm_thresh_kernel(
    const float* __restrict__ W,
    const uint8_t* __restrict__ act,
    int K,
    uint8_t* __restrict__ aT,
    float* __restrict__ aF) {
  __shared__ uint8_t ldsB[64 * 256];

  const int lane = threadIdx.x & 31;
  const int wave = threadIdx.x >> 5;
  const int n0   = (blockIdx.x * 4 + wave) * 16;
  const int b0   = blockIdx.y * 256;
  const int m    = lane & 15;
  const int g    = lane >> 4;

  v8i acc[16] = {};

  const float* wrow = W + (size_t)(n0 + m) * K;

  for (int kk = 0; kk < K; kk += 64) {
    __syncthreads();
#pragma unroll
    for (int i = 0; i < 8; ++i) {
      int slot = i * 128 + threadIdx.x;
      int row  = slot >> 4;
      int col  = (slot & 15) * 16;
#if USE_ASYNC_LDS
      __builtin_amdgcn_global_load_async_to_lds_b128(
          (as1_v4i*)(act + (size_t)(kk + row) * BATCH + b0 + col),
          (as3_v4i*)(ldsB + row * 256 + col), 0, 0);
#else
      *(int4*)(ldsB + row * 256 + col) =
          *(const int4*)(act + (size_t)(kk + row) * BATCH + b0 + col);
#endif
    }
#if USE_ASYNC_LDS
    __builtin_amdgcn_s_wait_asynccnt(0);
#endif
    __syncthreads();

    __builtin_prefetch(wrow + kk + 256, 0, 0);

    v8i A;
#pragma unroll
    for (int j = 0; j < 8; ++j) {
      int k0 = kk + g * 8 + (j & 1) * 4 + (j >> 1) * 16;
      const float4 w4 = *(const float4*)(wrow + k0);
      A[j] = pack4_i8(w4.x, w4.y, w4.z, w4.w);
    }

#pragma unroll
    for (int bt = 0; bt < 16; ++bt) {
      v8i Bv;
#pragma unroll
      for (int j = 0; j < 8; ++j) {
        const uint8_t* pb = ldsB + (g * 8 + (j & 1) * 4 + (j >> 1) * 16) * 256 + bt * 16 + m;
        Bv[j] = (int)((uint32_t)pb[0] | ((uint32_t)pb[256] << 8) |
                      ((uint32_t)pb[512] << 16) | ((uint32_t)pb[768] << 24));
      }
      acc[bt] = __builtin_amdgcn_wmma_i32_16x16x64_iu8(
          true, A, false, Bv, acc[bt], false, false);
    }
  }

#pragma unroll
  for (int bt = 0; bt < 16; ++bt) {
    int batch = b0 + bt * 16 + m;
#pragma unroll
    for (int r = 0; r < 8; ++r) {
      int neuron = n0 + r + 8 * g;
      uint8_t v  = (acc[bt][r] >= THRESH) ? 1 : 0;
      if (WRITE_T) aT[(size_t)neuron * BATCH + batch] = v;
      aF[(size_t)batch * HIDDEN + neuron] = (float)v;
    }
  }
}

__global__ __launch_bounds__(32) void out_gemm_kernel(
    const float* __restrict__ a1F, const float* __restrict__ a2F,
    const float* __restrict__ Wout, float* __restrict__ out) {
  const int lane = threadIdx.x & 31;
  const int m    = lane & 15;
  const int g    = lane >> 4;
  const int c0   = blockIdx.x * 16;
  const int b0   = blockIdx.y * 16;
  const int  n   = c0 + m;
  const bool nok = (n < CLASSES);
  const int  nc  = nok ? n : (CLASSES - 1);

  const float* p0 = Wout + (size_t)(2 * g) * CLASSES + nc;
  const float* p1 = p0 + (size_t)HIDDEN * CLASSES;

  v8f acc = {};
  const float* a1row = a1F + (size_t)(b0 + m) * HIDDEN;
  const float* a2row = a2F + (size_t)(b0 + m) * HIDDEN;

#pragma unroll 2
  for (int h = 0; h < HIDDEN; h += 4) {
    v2f A1 = *(const v2f*)(a1row + h + 2 * g);
    v2f A2 = *(const v2f*)(a2row + h + 2 * g);
    v2f B1, B2;
    B1.x = p0[(size_t)h * CLASSES];
    B1.y = p0[(size_t)(h + 1) * CLASSES];
    B2.x = p1[(size_t)h * CLASSES];
    B2.y = p1[(size_t)(h + 1) * CLASSES];
    acc = __builtin_amdgcn_wmma_f32_16x16x4_f32(false, A1, false, B1,
                                                (short)0, acc, false, false);
    acc = __builtin_amdgcn_wmma_f32_16x16x4_f32(false, A2, false, B2,
                                                (short)0, acc, false, false);
  }

  if (nok) {
#pragma unroll
    for (int r = 0; r < 8; ++r)
      out[(size_t)(b0 + r + 8 * g) * CLASSES + n] = acc[r];
  }
}

__global__ __launch_bounds__(128) void argmax_kernel(
    const float* __restrict__ out, int* __restrict__ pred) {
  int row  = blockIdx.x * 4 + (threadIdx.x >> 5);
  int lane = threadIdx.x & 31;
  const float* r = out + (size_t)row * CLASSES;
  float best = -__builtin_inff();
  int   bi   = 0x7fffffff;
  for (int c = lane; c < CLASSES; c += 32) {
    float v = r[c];
    if (v > best) { best = v; bi = c; }
  }
#pragma unroll
  for (int off = 16; off > 0; off >>= 1) {
    float ob = __shfl_xor(best, off, 32);
    int   oi = __shfl_xor(bi, off, 32);
    if (ob > best || (ob == best && oi < bi)) { best = ob; bi = oi; }
  }
  if (lane == 0) pred[row] = bi;
}

extern "C" void kernel_launch(void* const* d_in, const int* in_sizes, int n_in,
                              void* d_out, int out_size, void* d_ws,
                              size_t ws_size, hipStream_t stream) {
  const float* x    = (const float*)d_in[0];
  const float* W1   = (const float*)d_in[1];
  const float* W2   = (const float*)d_in[2];
  const float* Wout = (const float*)d_in[3];

  uint8_t* ws  = (uint8_t*)d_ws;
  uint8_t* a0T = ws;
  uint8_t* a1T = ws + (1u << 20);
  float*   a1F = (float*)(ws + (8u << 20));
  float*   a2F = (float*)(ws + (24u << 20));

  int*   pred = (int*)d_out;
  float* outp = (float*)d_out + BATCH;

  encode_kernel<<<dim3((BATCH * FEATURES + 255) / 256), 256, 0, stream>>>(x, a0T);

  dim3 gmm(HIDDEN / 16 / 4, BATCH / 256);
  binmm_thresh_kernel<true ><<<gmm, 128, 0, stream>>>(W1, a0T, ENC,    a1T,    a1F);
  binmm_thresh_kernel<false><<<gmm, 128, 0, stream>>>(W2, a1T, HIDDEN, nullptr, a2F);

  dim3 gout((CLASSES + 15) / 16, BATCH / 16);
  out_gemm_kernel<<<gout, 32, 0, stream>>>(a1F, a2F, Wout, outp);

  argmax_kernel<<<dim3(BATCH / 4), 128, 0, stream>>>(outp, pred);
}
